// MiniEfficientNetS4D_6356551598504
// MI455X (gfx1250) — matrix operand order repaired and checked by reading, not compiled, not run on hardware
//
#include <hip/hip_runtime.h>
#include <hip/hip_bf16.h>

// ============================================================================
// MiniEfficientNetS4D forward for MI455X (gfx1250, wave32, WMMA).
//
//  * Channel-last (NHWC) fp16 activations, channels padded to multiples of 32
//    -> every WMMA operand is a contiguous 16B/32B run per lane, no guards.
//  * 1x1 convs + stem (implicit GEMM, K=27 zero-padded to 32) use
//    v_wmma_f32_16x16x32_f16, one wave per 16(Co) x 16(pixels) tile.
//  * Depthwise / BN / SE / S4D head on VALU (tiny or irregular), all reads
//    coalesced in channel-last layout.
//  * S4D: rfft(2L)->mul->irfft[:L] == exact causal conv (L=32), fused with
//    GELU, 80x40 projection, GLU and readout MLP in one workgroup.
// ============================================================================

typedef __attribute__((ext_vector_type(16))) _Float16 v16h;
typedef __attribute__((ext_vector_type(8)))  _Float16 v8h;
typedef __attribute__((ext_vector_type(8)))  float    v8f;

#define IMG 64          // B*T images
#define EPS 1e-3f

// ---------------------------------------------------------------------------
// Input index map: setup_inputs() dict order ('x' first), params flattened
// with dict keys sorted (jax tree order), tuples (g,b) in order.
// ---------------------------------------------------------------------------
enum {
  IN_X = 0,
  B1_BN1_G, B1_BN1_B, B1_BN2_G, B1_BN2_B, B1_DW, B1_PW,
  B1_SE_B1, B1_SE_B2, B1_SE_W1, B1_SE_W2,
  B2A_BN0_G, B2A_BN0_B, B2A_BN1_G, B2A_BN1_B, B2A_BN2_G, B2A_BN2_B,
  B2A_DW, B2A_EXP, B2A_PW, B2A_SE_B1, B2A_SE_B2, B2A_SE_W1, B2A_SE_W2,
  B2B_BN0_G, B2B_BN0_B, B2B_BN1_G, B2B_BN1_B, B2B_BN2_G, B2B_BN2_B,
  B2B_DW, B2B_EXP, B2B_PW, B2B_SE_B1, B2B_SE_B2, B2B_SE_W1, B2B_SE_W2,
  B3A_BN0_G, B3A_BN0_B, B3A_BN1_G, B3A_BN1_B, B3A_BN2_G, B3A_BN2_B,
  B3A_DW, B3A_EXP, B3A_PW, B3A_SE_B1, B3A_SE_B2, B3A_SE_W1, B3A_SE_W2,
  B3B_BN0_G, B3B_BN0_B, B3B_BN1_G, B3B_BN1_B, B3B_BN2_G, B3B_BN2_B,
  B3B_DW, B3B_EXP, B3B_PW, B3B_SE_B1, B3B_SE_B2, B3B_SE_W1, B3B_SE_W2,
  RO_B1, RO_B2, RO_W1, RO_W2,
  S4D_C, S4D_D, S4D_LOGA, S4D_LOGDT, S4D_OUTB, S4D_OUTW,
  STEM_BN_G, STEM_BN_B, STEM_W
};

// ---------------------------------------------------------------------------
// 1x1 conv as WMMA GEMM, channel-last, fully guard-free.
//   X: [IMG][P][CinPad] f16, W: [CoPad][CinPad] f16 (zero padded),
//   Y: [IMG][P][CoPad] f16.
// CDNA5 operand layouts (ISA 7.12.2):
//   A lane m=l&15 needs K {kb..kb+7} and {kb+16..kb+23}, kb=(l>=16)*8
//   B lane n=l&15 needs K {kb..kb+7} and {kb+16..kb+23}, same as A
//   D lane n=l&15, VGPR v -> Co = v + (l>=16)*8        (contiguous store)
// ---------------------------------------------------------------------------
__global__ void conv1x1_wmma(const _Float16* __restrict__ X,
                             const _Float16* __restrict__ W,
                             _Float16* __restrict__ Y,
                             int CinPad, int CoPad, int P) {
  int img  = blockIdx.z;
  int m0   = blockIdx.y * 16;
  int n0   = blockIdx.x * 16;
  int lane = threadIdx.x;
  int r    = lane & 15;
  int kb   = (lane >> 4) * 8;
  int kh   = (lane >> 4) * 16;
  const _Float16* arow = W + (size_t)(m0 + r) * CinPad + kb;
  const _Float16* brow = X + ((size_t)img * P + (n0 + r)) * CinPad + kb;
  v8f acc = {};
  for (int k0 = 0; k0 < CinPad; k0 += 32) {
    v8h a_lo = *(const v8h*)(arow + k0);
    v8h a_hi = *(const v8h*)(arow + k0 + 16);
    v8h b_lo = *(const v8h*)(brow + k0);
    v8h b_hi = *(const v8h*)(brow + k0 + 16);
    v16h a, b;
#pragma unroll
    for (int e = 0; e < 8; ++e) {
      a[e] = a_lo[e]; a[8 + e] = a_hi[e];
      b[e] = b_lo[e]; b[8 + e] = b_hi[e];
    }
    acc = __builtin_amdgcn_wmma_f32_16x16x32_f16(false, a, false, b,
                                                 (short)0, acc, false, false);
  }
  v8h o;
#pragma unroll
  for (int v = 0; v < 8; ++v) o[v] = (_Float16)acc[v];
  *(v8h*)(Y + ((size_t)img * P + (n0 + r)) * CoPad + m0 + kb) = o;
}

// ---------------------------------------------------------------------------
// Stem conv 3x3 s2 p1 as implicit GEMM (K=27 zero-padded to 32, single WMMA).
// X: NCHW f32 input; W: [32][32] f16 padded; Y: [IMG][12544][32] f16 (NHWC).
// ---------------------------------------------------------------------------
__global__ void stem_conv_wmma(const float* __restrict__ X,
                               const _Float16* __restrict__ W,
                               _Float16* __restrict__ Y) {
  int img  = blockIdx.z;
  int m0   = blockIdx.y * 16;
  int n0   = blockIdx.x * 16;
  int lane = threadIdx.x;
  int r    = lane & 15;
  int kb   = (lane >> 4) * 8;
  int kh   = (lane >> 4) * 16;
  const _Float16* arow = W + (m0 + r) * 32;
  v8h a_lo = *(const v8h*)(arow + kb);
  v8h a_hi = *(const v8h*)(arow + kb + 16);
  v16h a;
#pragma unroll
  for (int e = 0; e < 8; ++e) { a[e] = a_lo[e]; a[8 + e] = a_hi[e]; }
  int p = n0 + r;
  int oy = p / 112, ox = p % 112;
  v16h b;
#pragma unroll
  for (int e = 0; e < 16; ++e) {
    int k = (e < 8) ? kb + e : 16 + kb + (e - 8);
    _Float16 val = (_Float16)0.f;
    if (k < 27) {
      int ci = k / 9, rem = k % 9, rr = rem / 3, ss = rem % 3;
      int iy = 2 * oy - 1 + rr, ix = 2 * ox - 1 + ss;
      if (iy >= 0 && iy < 224 && ix >= 0 && ix < 224)
        val = (_Float16)X[(((size_t)img * 3 + ci) * 224 + iy) * 224 + ix];
    }
    b[e] = val;
  }
  v8f acc = {};
  acc = __builtin_amdgcn_wmma_f32_16x16x32_f16(false, a, false, b,
                                               (short)0, acc, false, false);
  v8h o;
#pragma unroll
  for (int v = 0; v < 8; ++v) o[v] = (_Float16)acc[v];
  *(v8h*)(Y + ((size_t)img * 12544 + p) * 32 + m0 + kb) = o;
}

// ---------------------------------------------------------------------------
// Weight pad+cast: src f32 [Co][Cin] -> dst f16 [CoPad][CinPad], zeros in pad.
// ---------------------------------------------------------------------------
__global__ void pad_weight_f16(const float* __restrict__ src, _Float16* __restrict__ dst,
                               int Co, int Cin, int CoPad, int CinPad) {
  int i = blockIdx.x * blockDim.x + threadIdx.x;
  if (i >= CoPad * CinPad) return;
  int co = i / CinPad, ci = i % CinPad;
  dst[i] = (co < Co && ci < Cin) ? (_Float16)src[co * Cin + ci] : (_Float16)0.f;
}

// ---------------------------------------------------------------------------
// BatchNorm: stats (channel-last, coalesced) + apply (+SiLU, +residual).
// ---------------------------------------------------------------------------
__global__ void zero_f32(float* p, int n) {
  for (int i = threadIdx.x; i < n; i += blockDim.x) p[i] = 0.f;
}

__global__ void bn_stats(const _Float16* __restrict__ X, float* __restrict__ stats,
                         int C, int Cpad, int P, int chunk) {
  int img = blockIdx.x;
  int p0  = blockIdx.y * chunk;
  int c   = threadIdx.x;            // blockDim.x == Cpad
  if (c >= C) return;
  const _Float16* x = X + (size_t)img * P * Cpad;
  int pend = p0 + chunk; if (pend > P) pend = P;
  float s = 0.f, q = 0.f;
  for (int p = p0; p < pend; ++p) {
    float v = (float)x[(size_t)p * Cpad + c];
    s += v; q += v * v;
  }
  atomicAdd(&stats[c], s);
  atomicAdd(&stats[C + c], q);
}

__global__ void bn_apply(const _Float16* __restrict__ X, _Float16* __restrict__ Y,
                         const float* __restrict__ stats,
                         const float* __restrict__ g, const float* __restrict__ bt,
                         const _Float16* __restrict__ residual,
                         int C, int Cpad, int P, int act /*0=none,1=silu*/) {
  size_t i = (size_t)blockIdx.x * blockDim.x + threadIdx.x;
  size_t total = (size_t)IMG * P * Cpad;
  if (i >= total) return;
  int c = (int)(i % Cpad);
  if (c >= C) { Y[i] = (_Float16)0.f; return; }   // keep pad channels exactly 0
  float cnt = (float)IMG * (float)P;
  float m   = stats[c] / cnt;
  float var = stats[C + c] / cnt - m * m;
  float v = ((float)X[i] - m) * rsqrtf(var + EPS) * g[c] + bt[c];
  if (act == 1) v = v / (1.f + __expf(-v));
  if (residual) v += (float)residual[i];
  Y[i] = (_Float16)v;
}

// ---------------------------------------------------------------------------
// Depthwise KxK conv, channel-last (consecutive lanes = consecutive channels).
// ---------------------------------------------------------------------------
__global__ void dw_conv(const _Float16* __restrict__ X, const float* __restrict__ W,
                        _Float16* __restrict__ Y,
                        int C, int Cpad, int Hin, int Hout, int K, int stride) {
  size_t i = (size_t)blockIdx.x * blockDim.x + threadIdx.x;
  size_t total = (size_t)IMG * Hout * Hout * Cpad;
  if (i >= total) return;
  int c = (int)(i % Cpad);
  if (c >= C) { Y[i] = (_Float16)0.f; return; }
  int p   = (int)((i / Cpad) % ((size_t)Hout * Hout));
  int img = (int)(i / ((size_t)Cpad * Hout * Hout));
  int oy = p / Hout, ox = p % Hout;
  int pad = K / 2;
  const _Float16* x = X + (size_t)img * Hin * Hin * Cpad;
  const float*    w = W + (size_t)c * K * K;
  float acc = 0.f;
  for (int rr = 0; rr < K; ++rr) {
    int iy = oy * stride - pad + rr;
    if (iy < 0 || iy >= Hin) continue;
    for (int ss = 0; ss < K; ++ss) {
      int ix = ox * stride - pad + ss;
      if (ix < 0 || ix >= Hin) continue;
      acc += w[rr * K + ss] * (float)x[((size_t)iy * Hin + ix) * Cpad + c];
    }
  }
  Y[i] = (_Float16)acc;
}

// ---------------------------------------------------------------------------
// Squeeze-Excite (channel-last): means -> silu(W1 s+b1) -> sigmoid(W2 s1+b2).
// ---------------------------------------------------------------------------
__global__ void se_reduce(const _Float16* __restrict__ X,
                          const float* __restrict__ w1, const float* __restrict__ b1,
                          const float* __restrict__ w2, const float* __restrict__ b2,
                          float* __restrict__ sbuf, int C, int Cpad, int SQ, int P) {
  int img = blockIdx.x;
  __shared__ float s[256];
  __shared__ float s1[16];
  int t = threadIdx.x;
  if (t < C) {
    const _Float16* x = X + (size_t)img * P * Cpad;
    float acc = 0.f;
    for (int p = 0; p < P; ++p) acc += (float)x[(size_t)p * Cpad + t];
    s[t] = acc / (float)P;
  }
  __syncthreads();
  if (t < SQ) {
    float a = b1[t];
    for (int c = 0; c < C; ++c) a += w1[t * C + c] * s[c];
    s1[t] = a / (1.f + __expf(-a));
  }
  __syncthreads();
  if (t < C) {
    float a = b2[t];
    for (int q = 0; q < SQ; ++q) a += w2[t * SQ + q] * s1[q];
    sbuf[(size_t)img * C + t] = 1.f / (1.f + __expf(-a));
  }
}

__global__ void se_scale(_Float16* __restrict__ X, const float* __restrict__ sbuf,
                         int C, int Cpad, int P) {
  size_t i = (size_t)blockIdx.x * blockDim.x + threadIdx.x;
  size_t total = (size_t)IMG * P * Cpad;
  if (i >= total) return;
  int c   = (int)(i % Cpad);
  int img = (int)(i / ((size_t)P * Cpad));
  float sc = (c < C) ? sbuf[(size_t)img * C + c] : 0.f;  // pad stays 0
  X[i] = (_Float16)((float)X[i] * sc);
}

// ---------------------------------------------------------------------------
// Spatial sum -> sequence u[b][h][l].
// ---------------------------------------------------------------------------
__global__ void seq_sum(const _Float16* __restrict__ X, float* __restrict__ U,
                        int C, int Cpad, int P, int T) {
  int img = blockIdx.x;
  int b = img / T, l = img % T;
  int c = threadIdx.x;
  if (c < C) {
    const _Float16* x = X + (size_t)img * P * Cpad;
    float acc = 0.f;
    for (int p = 0; p < P; ++p) acc += (float)x[(size_t)p * Cpad + c];
    U[((size_t)b * C + c) * T + l] = acc;
  }
}

// ---------------------------------------------------------------------------
// Fused S4D head (exact causal conv == rfft/irfft path) + GELU + projection
// + GLU + readout MLP. Single workgroup, LDS resident (~37 KB << 320 KB WGP).
// ---------------------------------------------------------------------------
__global__ void s4d_readout(const float* __restrict__ U,
                            const float* __restrict__ Cm, const float* __restrict__ Dv,
                            const float* __restrict__ logA, const float* __restrict__ logdt,
                            const float* __restrict__ out_b, const float* __restrict__ out_w,
                            const float* __restrict__ ro_b1, const float* __restrict__ ro_b2,
                            const float* __restrict__ ro_w1, const float* __restrict__ ro_w2,
                            float* __restrict__ out) {
  const int H = 40, L = 32, N = 32, O = 80;
  __shared__ float kk[H * L];
  __shared__ float yy[2 * H * L];
  __shared__ float y2[2 * O * L];
  __shared__ float hf[2 * H];
  __shared__ float r1[2 * 64];
  int t = threadIdx.x;
  for (int i = t; i < H * L; i += blockDim.x) {
    int h = i / L, l = i % L;
    float dt = __expf(logdt[h]);
    float acc = 0.f;
    for (int n = 0; n < N; ++n) {
      float A   = -__expf(logA[h * N + n]);
      float dtA = A * dt;
      float Cc  = Cm[h * N + n] * (__expf(dtA) - 1.f) / A;
      acc += Cc * __expf(dtA * (float)l);
    }
    kk[i] = 2.f * acc;
  }
  __syncthreads();
  for (int i = t; i < 2 * H * L; i += blockDim.x) {
    int l = i % L, h = (i / L) % H, b = i / (H * L);
    const float* u = U + ((size_t)b * H + h) * L;
    float acc = Dv[h] * u[l];
    for (int m = 0; m <= l; ++m) acc += kk[h * L + m] * u[l - m];
    yy[i] = 0.5f * acc * (1.f + erff(acc * 0.70710678118f));
  }
  __syncthreads();
  for (int i = t; i < 2 * O * L; i += blockDim.x) {
    int l = i % L, o = (i / L) % O, b = i / (O * L);
    float acc = out_b[o];
    for (int h = 0; h < H; ++h) acc += out_w[o * H + h] * yy[(b * H + h) * L + l];
    y2[i] = acc;
  }
  __syncthreads();
  for (int i = t; i < 2 * H; i += blockDim.x) {
    int h = i % H, b = i / H;
    float a = y2[(b * O + h) * L + (L - 1)];
    float g = y2[(b * O + H + h) * L + (L - 1)];
    hf[i] = a / (1.f + __expf(-g));
  }
  __syncthreads();
  for (int i = t; i < 2 * 64; i += blockDim.x) {
    int j = i % 64, b = i / 64;
    float acc = ro_b1[j];
    for (int h = 0; h < H; ++h) acc += hf[b * H + h] * ro_w1[h * 64 + j];
    r1[i] = acc > 0.f ? acc : 0.f;
  }
  __syncthreads();
  for (int i = t; i < 2 * 60; i += blockDim.x) {
    int k = i % 60, b = i / 60;
    float acc = ro_b2[k];
    for (int j = 0; j < 64; ++j) acc += r1[b * 64 + j] * ro_w2[j * 60 + k];
    out[b * 60 + k] = acc;
  }
}

// ---------------------------------------------------------------------------
// Host orchestration (graph-capture safe: stream-ordered launches only).
// ---------------------------------------------------------------------------
static inline int cdiv(int a, int b) { return (a + b - 1) / b; }
static inline int cpad32(int c) { return (c + 31) & ~31; }

// Workspace: channel-last fp16 activation regions (~270 MB total).
static const size_t SZ_A  = (size_t)IMG * 12544 * 96  * 2;  // 154.1 MB
static const size_t SZ_B  = (size_t)IMG * 3136  * 160 * 2;  //  64.2 MB
static const size_t SZ_C  = (size_t)IMG * 12544 * 32  * 2;  //  51.4 MB
static const size_t OFF_A = 0;
static const size_t OFF_B = OFF_A + SZ_A;
static const size_t OFF_C = OFF_B + SZ_B;
static const size_t OFF_STATS = OFF_C + SZ_C;               // 2*256 f32
static const size_t OFF_SBUF  = OFF_STATS + 2048;           // 64*256 f32
static const size_t OFF_U     = OFF_SBUF + 64 * 256 * 4;    // 2*40*32 f32
static const size_t OFF_W16   = OFF_U + 16384;              // padded f16 weights

extern "C" void kernel_launch(void* const* d_in, const int* in_sizes, int n_in,
                              void* d_out, int out_size, void* d_ws, size_t ws_size,
                              hipStream_t stream) {
  (void)in_sizes; (void)n_in; (void)out_size; (void)ws_size;
  auto F = [&](int i) { return (const float*)d_in[i]; };
  char* ws = (char*)d_ws;
  _Float16* bufA = (_Float16*)(ws + OFF_A);
  _Float16* bufB = (_Float16*)(ws + OFF_B);
  _Float16* bufC = (_Float16*)(ws + OFF_C);
  float* stats = (float*)(ws + OFF_STATS);
  float* sbuf  = (float*)(ws + OFF_SBUF);
  float* u     = (float*)(ws + OFF_U);
  _Float16* w16 = (_Float16*)(ws + OFF_W16);

  // Padded f16 weight slots (element offsets; [CoPad][CinPad] each).
  _Float16* wStem  = w16 + 0;      // 32 x 32
  _Float16* wPw1   = w16 + 1024;   // 32 x 32   (16x32)
  _Float16* wPw2a  = w16 + 2048;   // 32 x 96   (24x96)
  _Float16* wPw2b  = w16 + 5120;   // 32 x 160  (24x144)
  _Float16* wPw3a  = w16 + 10240;  // 64 x 160  (40x144)
  _Float16* wPw3b  = w16 + 20480;  // 64 x 256  (40x240)
  _Float16* wExp2a = w16 + 36864;  // 96 x 32   (96x16)
  _Float16* wExp2b = w16 + 39936;  // 160 x 32  (144x24)
  _Float16* wExp3a = w16 + 45056;  // 160 x 32  (144x24)
  _Float16* wExp3b = w16 + 50176;  // 256 x 64  (240x40)

  auto padw = [&](int idx, _Float16* dst, int Co, int Cin) {
    int CoP = cpad32(Co), CinP = cpad32(Cin);
    pad_weight_f16<<<cdiv(CoP * CinP, 256), 256, 0, stream>>>(F(idx), dst,
                                                              Co, Cin, CoP, CinP);
  };
  padw(STEM_W,  wStem,  32, 27);
  padw(B1_PW,   wPw1,   16, 32);
  padw(B2A_PW,  wPw2a,  24, 96);
  padw(B2B_PW,  wPw2b,  24, 144);
  padw(B3A_PW,  wPw3a,  40, 144);
  padw(B3B_PW,  wPw3b,  40, 240);
  padw(B2A_EXP, wExp2a, 96, 16);
  padw(B2B_EXP, wExp2b, 144, 24);
  padw(B3A_EXP, wExp3a, 144, 24);
  padw(B3B_EXP, wExp3b, 240, 40);

  auto batchnorm = [&](const _Float16* xin, _Float16* xout, int gi, int bi,
                       int C, int P, int act, const _Float16* res) {
    int Cpad = cpad32(C);
    zero_f32<<<1, 256, 0, stream>>>(stats, 2 * C);
    bn_stats<<<dim3(IMG, cdiv(P, 512)), Cpad, 0, stream>>>(xin, stats, C, Cpad, P, 512);
    size_t tot = (size_t)IMG * P * Cpad;
    bn_apply<<<cdiv((int)tot, 256), 256, 0, stream>>>(xin, xout, stats,
                                                      F(gi), F(bi), res, C, Cpad, P, act);
  };
  auto pconv = [&](const _Float16* X, const _Float16* W, _Float16* Y,
                   int Cin, int Co, int P) {
    int CinP = cpad32(Cin), CoP = cpad32(Co);
    conv1x1_wmma<<<dim3(P / 16, CoP / 16, IMG), 32, 0, stream>>>(X, W, Y, CinP, CoP, P);
  };
  auto dconv = [&](const _Float16* X, int wi, _Float16* Y,
                   int C, int Hin, int Hout, int K, int s) {
    int Cpad = cpad32(C);
    size_t tot = (size_t)IMG * Hout * Hout * Cpad;
    dw_conv<<<cdiv((int)tot, 256), 256, 0, stream>>>(X, F(wi), Y, C, Cpad,
                                                     Hin, Hout, K, s);
  };
  auto se = [&](_Float16* X, int w1i, int b1i, int w2i, int b2i,
                int C, int SQ, int P) {
    int Cpad = cpad32(C);
    se_reduce<<<IMG, 256, 0, stream>>>(X, F(w1i), F(b1i), F(w2i), F(b2i),
                                       sbuf, C, Cpad, SQ, P);
    size_t tot = (size_t)IMG * P * Cpad;
    se_scale<<<cdiv((int)tot, 256), 256, 0, stream>>>(X, sbuf, C, Cpad, P);
  };

  const int P1 = 112 * 112, P2 = 56 * 56, P3 = 28 * 28;

  // ---- stem: 3x224^2 -> 32@112^2 (WMMA implicit GEMM) ----
  stem_conv_wmma<<<dim3(784, 2, IMG), 32, 0, stream>>>((const float*)d_in[IN_X],
                                                       wStem, bufA);
  batchnorm(bufA, bufA, STEM_BN_G, STEM_BN_B, 32, P1, 1, nullptr);

  // ---- b1 (32 -> 16) ----
  dconv(bufA, B1_DW, bufB, 32, 112, 112, 3, 1);
  batchnorm(bufB, bufB, B1_BN1_G, B1_BN1_B, 32, P1, 1, nullptr);
  se(bufB, B1_SE_W1, B1_SE_B1, B1_SE_W2, B1_SE_B2, 32, 8, P1);
  pconv(bufB, wPw1, bufC, 32, 16, P1);
  batchnorm(bufC, bufC, B1_BN2_G, B1_BN2_B, 16, P1, 0, nullptr);

  // ---- b2a (16 -> 96 -> 24, s2) ----
  pconv(bufC, wExp2a, bufA, 16, 96, P1);
  batchnorm(bufA, bufA, B2A_BN0_G, B2A_BN0_B, 96, P1, 1, nullptr);
  dconv(bufA, B2A_DW, bufB, 96, 112, 56, 3, 2);
  batchnorm(bufB, bufB, B2A_BN1_G, B2A_BN1_B, 96, P2, 1, nullptr);
  se(bufB, B2A_SE_W1, B2A_SE_B1, B2A_SE_W2, B2A_SE_B2, 96, 4, P2);
  pconv(bufB, wPw2a, bufC, 96, 24, P2);
  batchnorm(bufC, bufC, B2A_BN2_G, B2A_BN2_B, 24, P2, 0, nullptr);

  // ---- b2b (24 -> 144 -> 24, residual) ----
  pconv(bufC, wExp2b, bufA, 24, 144, P2);
  batchnorm(bufA, bufA, B2B_BN0_G, B2B_BN0_B, 144, P2, 1, nullptr);
  dconv(bufA, B2B_DW, bufB, 144, 56, 56, 3, 1);
  batchnorm(bufB, bufB, B2B_BN1_G, B2B_BN1_B, 144, P2, 1, nullptr);
  se(bufB, B2B_SE_W1, B2B_SE_B1, B2B_SE_W2, B2B_SE_B2, 144, 6, P2);
  pconv(bufB, wPw2b, bufA, 144, 24, P2);
  batchnorm(bufA, bufA, B2B_BN2_G, B2B_BN2_B, 24, P2, 0, bufC);  // + residual

  // ---- b3a (24 -> 144 -> 40, k=5, s2) ----
  pconv(bufA, wExp3a, bufB, 24, 144, P2);
  batchnorm(bufB, bufB, B3A_BN0_G, B3A_BN0_B, 144, P2, 1, nullptr);
  dconv(bufB, B3A_DW, bufC, 144, 56, 28, 5, 2);
  batchnorm(bufC, bufC, B3A_BN1_G, B3A_BN1_B, 144, P3, 1, nullptr);
  se(bufC, B3A_SE_W1, B3A_SE_B1, B3A_SE_W2, B3A_SE_B2, 144, 6, P3);
  pconv(bufC, wPw3a, bufA, 144, 40, P3);
  batchnorm(bufA, bufA, B3A_BN2_G, B3A_BN2_B, 40, P3, 0, nullptr);

  // ---- b3b (40 -> 240 -> 40, k=5, residual) ----
  pconv(bufA, wExp3b, bufB, 40, 240, P3);
  batchnorm(bufB, bufB, B3B_BN0_G, B3B_BN0_B, 240, P3, 1, nullptr);
  dconv(bufB, B3B_DW, bufC, 240, 28, 28, 5, 1);
  batchnorm(bufC, bufC, B3B_BN1_G, B3B_BN1_B, 240, P3, 1, nullptr);
  se(bufC, B3B_SE_W1, B3B_SE_B1, B3B_SE_W2, B3B_SE_B2, 240, 10, P3);
  pconv(bufC, wPw3b, bufB, 240, 40, P3);
  batchnorm(bufB, bufB, B3B_BN2_G, B3B_BN2_B, 40, P3, 0, bufA);  // + residual

  // ---- sequence head ----
  seq_sum<<<IMG, 64, 0, stream>>>(bufB, u, 40, 64, P3, 32);
  s4d_readout<<<1, 256, 0, stream>>>(u, F(S4D_C), F(S4D_D), F(S4D_LOGA),
                                     F(S4D_LOGDT), F(S4D_OUTB), F(S4D_OUTW),
                                     F(RO_B1), F(RO_B2), F(RO_W1), F(RO_W2),
                                     (float*)d_out);
}
